// kv_cache_multihead_20495583936573
// MI455X (gfx1250) — matrix operand order repaired and checked by reading, not compiled, not run on hardware
//
#include <hip/hip_runtime.h>
#include <hip/hip_bf16.h>

// ---------------------------------------------------------------------------
// Problem constants
// ---------------------------------------------------------------------------
#define EMBED   2048
#define HEADS   16
#define HDIM    128
#define SEQ     1024
#define BATCH   4
#define MROWS   (BATCH * SEQ)                 // 4096
#define ATT_SCALE 0.08838834764831845f        // 1/sqrt(128)

typedef __attribute__((ext_vector_type(16))) _Float16 v16h;
typedef __attribute__((ext_vector_type(8)))  _Float16 v8h;
typedef __attribute__((ext_vector_type(8)))  float    v8f;

// ---------------------------------------------------------------------------
// Async global -> LDS copy (gfx1250 GLOBAL_LOAD_ASYNC_TO_LDS_B128, ASYNCcnt).
// The builtin's probed signature is:
//   void __builtin_amdgcn_global_load_async_to_lds_b128(
//       int4 __device__* src, int4 __shared__* dst, imm int offset, imm cpol)
// where int4 is a GCC-style vector_size(16) int vector.
// ---------------------------------------------------------------------------
#if __has_builtin(__builtin_amdgcn_global_load_async_to_lds_b128)
#define HAVE_ASYNC_LDS 1
#else
#define HAVE_ASYNC_LDS 0
#endif

typedef int b128_t __attribute__((vector_size(16)));
typedef __attribute__((address_space(1))) b128_t* gas_b128_p;  // global (AS1)
typedef __attribute__((address_space(3))) b128_t* las_b128_p;  // LDS (AS3)

__device__ __forceinline__ void async_copy16(const _Float16* g, _Float16* l) {
#if HAVE_ASYNC_LDS
  __builtin_amdgcn_global_load_async_to_lds_b128(
      (gas_b128_p)(void*)const_cast<_Float16*>(g),
      (las_b128_p)(void*)l,
      /*offset=*/0, /*cpol=*/0);
#else
  *(v8h*)l = *(const v8h*)g;
#endif
}

template <int N>
__device__ __forceinline__ void wait_async() {
#if __has_builtin(__builtin_amdgcn_s_wait_asynccnt)
  __builtin_amdgcn_s_wait_asynccnt((unsigned short)N);
#else
  asm volatile("s_wait_asynccnt %0" ::"i"(N) : "memory");
#endif
}

// ---------------------------------------------------------------------------
// WMMA operand loaders (wave32, 16x16x32 f16 shapes, CDNA5 VGPR layouts)
//   A 16x32 (MxK): lane L -> row M = L&15; halfs 0..7 = K[8*(L>>4)..+7],
//                  halfs 8..15 = K[16+8*(L>>4)..+7]
//   B 32x16 (KxN): lane L -> col N = L&15; halfs 0..7 = K[8*(L>>4)..+7],
//                  halfs 8..15 = K[16+8*(L>>4)..+7]
// Sources are row-major [row, k] with leading-dim `ld` (halfs), 16B-aligned.
// ---------------------------------------------------------------------------
__device__ __forceinline__ v16h cat16(v8h lo, v8h hi) {
  v16h r;
#pragma unroll
  for (int i = 0; i < 8; ++i) { r[i] = lo[i]; r[i + 8] = hi[i]; }
  return r;
}

__device__ __forceinline__ v16h load_opA(const _Float16* __restrict__ src, int ld) {
  const int lane = threadIdx.x & 31;
  const int m = lane & 15, hh = lane >> 4;
  const _Float16* p = src + m * ld + 8 * hh;
  return cat16(*(const v8h*)p, *(const v8h*)(p + 16));
}

__device__ __forceinline__ v16h load_opB(const _Float16* __restrict__ src, int ld) {
  const int lane = threadIdx.x & 31;
  const int n = lane & 15, hh = lane >> 4;
  const _Float16* p = src + n * ld + 8 * hh;
  return cat16(*(const v8h*)p, *(const v8h*)(p + 16));
}

__device__ __forceinline__ v8f wmma16(v16h a, v16h b, v8f c) {
  return __builtin_amdgcn_wmma_f32_16x16x32_f16(
      /*neg_a=*/false, a, /*neg_b=*/false, b,
      /*c_mod=*/(short)0, c, /*reuse_a=*/false, /*reuse_b=*/false);
}

// ---------------------------------------------------------------------------
// fp32 -> fp16 conversion
// ---------------------------------------------------------------------------
__global__ __launch_bounds__(256) void f32_to_f16_kernel(
    const float* __restrict__ src, _Float16* __restrict__ dst, long n) {
  long i = (long)blockIdx.x * 256 + threadIdx.x;
  if (i < n) dst[i] = (_Float16)src[i];
}

// ---------------------------------------------------------------------------
// 128x128 GEMM core: C = A[M x 2048] * W[N x 2048]^T  (both K-contiguous)
// 256 threads = 8 waves; wave (wm, wn) owns 32 rows x 64 cols = 2x4 C tiles.
// Double-buffered LDS; chunk k+1 async-DMA overlaps chunk k WMMAs.
// ---------------------------------------------------------------------------
#define GLDA 40  // 32 + 8 pad halfs; row pitch 80B keeps 16B alignment
#define NKCH (EMBED / 32)

__device__ __forceinline__ void gemm_core_128x128(
    const _Float16* __restrict__ A, const _Float16* __restrict__ W,
    int m0, int n0, v8f c[2][4]) {
  __shared__ __attribute__((aligned(16))) _Float16 As[2][128 * GLDA];
  __shared__ __attribute__((aligned(16))) _Float16 Bs[2][128 * GLDA];
  const int tid = threadIdx.x;
  const int w = tid >> 5, wm = w & 3, wn = w >> 2;

#pragma unroll
  for (int i = 0; i < 2; ++i)
#pragma unroll
    for (int j = 0; j < 4; ++j)
#pragma unroll
      for (int r = 0; r < 8; ++r) c[i][j][r] = 0.0f;

  // 4 async b128 ops per thread per stage (2 for A panel, 2 for B panel).
  auto stage = [&](int buf, int k0) {
#pragma unroll
    for (int i = 0; i < 2; ++i) {
      int cid = tid + 256 * i;            // 512 chunks of 8 halfs per panel
      int row = cid >> 2;
      int ko  = (cid & 3) * 8;
      async_copy16(&A[(size_t)(m0 + row) * EMBED + k0 + ko],
                   &As[buf][row * GLDA + ko]);
      async_copy16(&W[(size_t)(n0 + row) * EMBED + k0 + ko],
                   &Bs[buf][row * GLDA + ko]);
    }
  };

  stage(0, 0);
  for (int kc = 0; kc < NKCH; ++kc) {
    const int cur = kc & 1;
    if (kc + 1 < NKCH) {
      stage(cur ^ 1, (kc + 1) * 32);      // prefetch next chunk (async)
      wait_async<4>();                    // current chunk done; next in flight
    } else {
      wait_async<0>();
    }
    __syncthreads();

    v16h a0 = load_opA(&As[cur][(wm * 32) * GLDA], GLDA);
    v16h a1 = load_opA(&As[cur][(wm * 32 + 16) * GLDA], GLDA);
    v16h b0 = load_opB(&Bs[cur][(wn * 64 + 0) * GLDA], GLDA);
    v16h b1 = load_opB(&Bs[cur][(wn * 64 + 16) * GLDA], GLDA);
    v16h b2 = load_opB(&Bs[cur][(wn * 64 + 32) * GLDA], GLDA);
    v16h b3 = load_opB(&Bs[cur][(wn * 64 + 48) * GLDA], GLDA);
    c[0][0] = wmma16(a0, b0, c[0][0]);
    c[1][0] = wmma16(a1, b0, c[1][0]);
    c[0][1] = wmma16(a0, b1, c[0][1]);
    c[1][1] = wmma16(a1, b1, c[1][1]);
    c[0][2] = wmma16(a0, b2, c[0][2]);
    c[1][2] = wmma16(a1, b2, c[1][2]);
    c[0][3] = wmma16(a0, b3, c[0][3]);
    c[1][3] = wmma16(a1, b3, c[1][3]);

    __syncthreads();  // all waves done reading `cur` before it is re-staged
  }
}

// ---------------------------------------------------------------------------
// QKV projection: z = 0/1/2 -> Q / K / V^T
//   Q, K stored (B,H,T,D) f16 ; V stored transposed (B,H,D,S) f16
// ---------------------------------------------------------------------------
__global__ __launch_bounds__(256) void qkv_gemm_kernel(
    const _Float16* __restrict__ xh,
    const _Float16* __restrict__ wqh, const _Float16* __restrict__ wkh,
    const _Float16* __restrict__ wvh,
    _Float16* __restrict__ Qb, _Float16* __restrict__ Kb,
    _Float16* __restrict__ VTb) {
  const int n0 = blockIdx.x * 128, m0 = blockIdx.y * 128, z = blockIdx.z;
  const _Float16* W = (z == 0) ? wqh : (z == 1) ? wkh : wvh;
  v8f c[2][4];
  gemm_core_128x128(xh, W, m0, n0, c);

  const int lane = threadIdx.x & 31, w = threadIdx.x >> 5;
  const int wm = w & 3, wn = w >> 2, hh = lane >> 4, nn = lane & 15;
#pragma unroll
  for (int mt = 0; mt < 2; ++mt)
#pragma unroll
    for (int nt = 0; nt < 4; ++nt)
#pragma unroll
      for (int r = 0; r < 8; ++r) {
        int m = m0 + wm * 32 + mt * 16 + r + 8 * hh;   // b*SEQ + t
        int n = n0 + wn * 64 + nt * 16 + nn;           // h*HDIM + d
        int b = m >> 10, t = m & (SEQ - 1);
        int h = n >> 7, d = n & (HDIM - 1);
        size_t bh = (size_t)(b * HEADS + h);
        _Float16 val = (_Float16)c[mt][nt][r];
        if (z == 2)      VTb[(bh * HDIM + d) * SEQ + t] = val;
        else if (z == 1) Kb[(bh * SEQ + t) * HDIM + d]  = val;
        else             Qb[(bh * SEQ + t) * HDIM + d]  = val;
      }
}

// ---------------------------------------------------------------------------
// RoPE (reference bug preserved: angle depends on HEAD index, not position)
// Layout (B,H,T,D): row = bh*SEQ + t ; pair j in [0,64)
// ---------------------------------------------------------------------------
__global__ __launch_bounds__(256) void rope_kernel(
    _Float16* __restrict__ Qb, _Float16* __restrict__ Kb, long npairs) {
  long e = (long)blockIdx.x * 256 + threadIdx.x;
  if (e >= npairs) return;
  int  j   = (int)(e & 63);
  long row = e >> 6;                       // bh*SEQ + t
  int  h   = (int)((row >> 10) & (HEADS - 1));
  float inv = __powf(10000.0f, -(float)j / 64.0f);
  float ang = (float)h * inv;
  float cs = __cosf(ang), sn = __sinf(ang);
  long base = row * HDIM + 2 * j;
  float q0 = (float)Qb[base], q1 = (float)Qb[base + 1];
  Qb[base]     = (_Float16)(q0 * cs - q1 * sn);
  Qb[base + 1] = (_Float16)(q0 * sn + q1 * cs);
  float k0 = (float)Kb[base], k1 = (float)Kb[base + 1];
  Kb[base]     = (_Float16)(k0 * cs - k1 * sn);
  Kb[base + 1] = (_Float16)(k0 * sn + k1 * cs);
}

// ---------------------------------------------------------------------------
// Flash attention: block = (t-tile of 64, bh); 4 waves, wave owns 16 t-rows.
// Q preloaded in registers (A layout); per 64-wide s-chunk async-stage K and
// V^T into LDS, score WMMAs, online softmax, P via per-wave LDS re-layout,
// PV WMMAs.
// ---------------------------------------------------------------------------
#define KLDA 136   // 128 + 8 pad halfs
#define VLDA 72    // 64 + 8 pad halfs
#define PLDA 72

__global__ __launch_bounds__(128) void attn_kernel(
    const _Float16* __restrict__ Qb, const _Float16* __restrict__ Kb,
    const _Float16* __restrict__ VTb, _Float16* __restrict__ CTX) {
  __shared__ __attribute__((aligned(16))) _Float16 Ks[64 * KLDA];
  __shared__ __attribute__((aligned(16))) _Float16 Vs[HDIM * VLDA];
  __shared__ __attribute__((aligned(16))) _Float16 Ps[4][16 * PLDA];

  const int tblk = blockIdx.x;             // 0..15 (t-tile of 64)
  const int bh   = blockIdx.y;             // 0..63
  const int tid  = threadIdx.x, w = tid >> 5, lane = tid & 31;
  const int hh = lane >> 4, nn = lane & 15;
  const int t0 = tblk * 64 + w * 16;       // first t-row of this wave

  // Preload Q A-operands for all 4 K(=d)-chunks directly from global.
  const _Float16* Qbase = Qb + ((size_t)bh * SEQ + t0) * HDIM;
  v16h qa[4];
#pragma unroll
  for (int dk = 0; dk < 4; ++dk) qa[dk] = load_opA(Qbase + dk * 32, HDIM);

  v8f cacc[8];
#pragma unroll
  for (int dt = 0; dt < 8; ++dt)
#pragma unroll
    for (int r = 0; r < 8; ++r) cacc[dt][r] = 0.0f;
  float mrow[8], lrow[8];
#pragma unroll
  for (int r = 0; r < 8; ++r) { mrow[r] = -1e30f; lrow[r] = 0.0f; }

  const int nchunks = tblk + 1;            // causal: s-chunks up to diagonal
  for (int sc = 0; sc < nchunks; ++sc) {
    const int s0 = sc * 64;
    __syncthreads();                       // previous readers done
    // Async-stage K chunk (64 x 128) and V^T chunk (128 x 64) into LDS.
#pragma unroll
    for (int i = 0; i < 8; ++i) {
      int cid = tid + 128 * i;             // 1024 chunks of 8 halfs each
      int krow = cid >> 4, ko = (cid & 15) * 8;
      async_copy16(&Kb[((size_t)bh * SEQ + s0 + krow) * HDIM + ko],
                   &Ks[krow * KLDA + ko]);
      int vrow = cid >> 3, so = (cid & 7) * 8;
      async_copy16(&VTb[((size_t)bh * HDIM + vrow) * SEQ + s0 + so],
                   &Vs[vrow * VLDA + so]);
    }
    wait_async<0>();
    __syncthreads();

    // Scores: 16 t-rows x 64 s-cols, accumulate over d in 4 chunks of 32.
    v8f sacc[4];
#pragma unroll
    for (int st = 0; st < 4; ++st) {
#pragma unroll
      for (int r = 0; r < 8; ++r) sacc[st][r] = 0.0f;
#pragma unroll
      for (int dk = 0; dk < 4; ++dk)
        sacc[st] = wmma16(qa[dk], load_opB(&Ks[(st * 16) * KLDA + dk * 32], KLDA),
                          sacc[st]);
    }

    // Scale + causal mask + per-row max (C layout: row = r+8*hh, col = nn).
    float rmax[8];
#pragma unroll
    for (int r = 0; r < 8; ++r) rmax[r] = -1e30f;
#pragma unroll
    for (int st = 0; st < 4; ++st)
#pragma unroll
      for (int r = 0; r < 8; ++r) {
        int s = s0 + st * 16 + nn;
        int t = t0 + r + 8 * hh;
        float v = sacc[st][r] * ATT_SCALE;
        if (s > t) v = -1e30f;
        sacc[st][r] = v;
        rmax[r] = fmaxf(rmax[r], v);
      }
#pragma unroll
    for (int r = 0; r < 8; ++r)
#pragma unroll
      for (int off = 1; off < 16; off <<= 1)
        rmax[r] = fmaxf(rmax[r], __shfl_xor(rmax[r], off, 32));

    // Online softmax update.
    float alpha[8];
#pragma unroll
    for (int r = 0; r < 8; ++r) {
      float mnew = fmaxf(mrow[r], rmax[r]);
      alpha[r] = __expf(mrow[r] - mnew);
      mrow[r] = mnew;
    }
#pragma unroll
    for (int dt = 0; dt < 8; ++dt)
#pragma unroll
      for (int r = 0; r < 8; ++r) cacc[dt][r] *= alpha[r];

    float rsum[8];
#pragma unroll
    for (int r = 0; r < 8; ++r) rsum[r] = 0.0f;
#pragma unroll
    for (int st = 0; st < 4; ++st)
#pragma unroll
      for (int r = 0; r < 8; ++r) {
        float p = __expf(sacc[st][r] - mrow[r]);
        rsum[r] += p;
        Ps[w][(r + 8 * hh) * PLDA + st * 16 + nn] = (_Float16)p;  // C -> A relayout
      }
#pragma unroll
    for (int r = 0; r < 8; ++r) {
#pragma unroll
      for (int off = 1; off < 16; off <<= 1)
        rsum[r] += __shfl_xor(rsum[r], off, 32);
      lrow[r] = lrow[r] * alpha[r] + rsum[r];
    }

    // P(16x64) @ V(64x128): DS ops are in-order per wave; Ps is wave-private.
#pragma unroll
    for (int pc = 0; pc < 2; ++pc) {
      v16h pa = load_opA(&Ps[w][pc * 32], PLDA);
#pragma unroll
      for (int dt = 0; dt < 8; ++dt)
        cacc[dt] = wmma16(pa, load_opB(&Vs[(dt * 16) * VLDA + pc * 32], VLDA),
                          cacc[dt]);
    }
  }

  // Normalize and write context as (B,T,HEADS*HDIM) f16 for the out-proj GEMM.
  const int b = bh >> 4, h = bh & (HEADS - 1);
#pragma unroll
  for (int r = 0; r < 8; ++r) {
    float inv = 1.0f / lrow[r];
    int t = t0 + r + 8 * hh;
    size_t rowbase = ((size_t)b * SEQ + t) * EMBED + h * HDIM;
#pragma unroll
    for (int dt = 0; dt < 8; ++dt)
      CTX[rowbase + dt * 16 + nn] = (_Float16)(cacc[dt][r] * inv);
  }
}

// ---------------------------------------------------------------------------
// Output projection: out = ctx @ wo^T  (fp32 result straight to d_out)
// ---------------------------------------------------------------------------
__global__ __launch_bounds__(256) void proj_gemm_kernel(
    const _Float16* __restrict__ ctx, const _Float16* __restrict__ woh,
    float* __restrict__ out) {
  const int n0 = blockIdx.x * 128, m0 = blockIdx.y * 128;
  v8f c[2][4];
  gemm_core_128x128(ctx, woh, m0, n0, c);

  const int lane = threadIdx.x & 31, w = threadIdx.x >> 5;
  const int wm = w & 3, wn = w >> 2, hh = lane >> 4, nn = lane & 15;
#pragma unroll
  for (int mt = 0; mt < 2; ++mt)
#pragma unroll
    for (int nt = 0; nt < 4; ++nt)
#pragma unroll
      for (int r = 0; r < 8; ++r) {
        int m = m0 + wm * 32 + mt * 16 + r + 8 * hh;
        int n = n0 + wn * 64 + nt * 16 + nn;
        out[(size_t)m * EMBED + n] = c[mt][nt][r];
      }
}

// ---------------------------------------------------------------------------
// Host launcher
// ---------------------------------------------------------------------------
extern "C" void kernel_launch(void* const* d_in, const int* in_sizes, int n_in,
                              void* d_out, int out_size, void* d_ws, size_t ws_size,
                              hipStream_t stream) {
  const float* x  = (const float*)d_in[0];
  const float* wq = (const float*)d_in[1];
  const float* wk = (const float*)d_in[2];
  const float* wv = (const float*)d_in[3];
  const float* wo = (const float*)d_in[4];
  // d_in[5] = start_pos (== 0 in the reference setup)

  const long NX = (long)MROWS * EMBED;   // 8,388,608 activations
  const long NW = (long)EMBED * EMBED;   // 4,194,304 per weight

  _Float16* p   = (_Float16*)d_ws;
  _Float16* xh  = p;                     // [MROWS, EMBED]
  _Float16* wqh = xh  + NX;
  _Float16* wkh = wqh + NW;
  _Float16* wvh = wkh + NW;
  _Float16* woh = wvh + NW;
  _Float16* Qb  = woh + NW;              // (B,H,T,D)
  _Float16* Kb  = Qb  + NX;              // (B,H,S,D)
  _Float16* VTb = Kb  + NX;              // (B,H,D,S)
  _Float16* CTX = VTb + NX;              // (B,T,H*D)

  // 1) fp32 -> fp16
  f32_to_f16_kernel<<<(unsigned)(NX / 256), 256, 0, stream>>>(x, xh, NX);
  f32_to_f16_kernel<<<(unsigned)(NW / 256), 256, 0, stream>>>(wq, wqh, NW);
  f32_to_f16_kernel<<<(unsigned)(NW / 256), 256, 0, stream>>>(wk, wkh, NW);
  f32_to_f16_kernel<<<(unsigned)(NW / 256), 256, 0, stream>>>(wv, wvh, NW);
  f32_to_f16_kernel<<<(unsigned)(NW / 256), 256, 0, stream>>>(wo, woh, NW);

  // 2) Q/K/V projections (z picks the weight; V written transposed)
  dim3 gqkv(EMBED / 128, MROWS / 128, 3);
  qkv_gemm_kernel<<<gqkv, 256, 0, stream>>>(xh, wqh, wkh, wvh, Qb, Kb, VTb);

  // 3) RoPE on Q and K (head-indexed angles, matching the reference)
  const long npairs = (long)BATCH * HEADS * SEQ * (HDIM / 2);  // 4,194,304
  rope_kernel<<<(unsigned)((npairs + 255) / 256), 256, 0, stream>>>(Qb, Kb, npairs);

  // 4) Causal flash attention
  dim3 gattn(SEQ / 64, BATCH * HEADS);
  attn_kernel<<<gattn, 128, 0, stream>>>(Qb, Kb, VTb, CTX);

  // 5) Output projection -> fp32 d_out
  dim3 gproj(EMBED / 128, MROWS / 128);
  proj_gemm_kernel<<<gproj, 256, 0, stream>>>(CTX, woh, (float*)d_out);
}
